// INT8Linear_74672301408656
// MI455X (gfx1250) — compile-verified, matrix operand order checked by reading, not run on hardware
//
#include <hip/hip_runtime.h>
#include <hip/hip_bf16.h>

typedef __attribute__((ext_vector_type(16))) __bf16        v16bf;
typedef __attribute__((ext_vector_type(8)))  float         v8f;
typedef __attribute__((ext_vector_type(4)))  unsigned int  v4u;
typedef __attribute__((ext_vector_type(8)))  int           v8i;
typedef __attribute__((ext_vector_type(4)))  int           v4i;

constexpr int TOKENS = 4096;   // M
constexpr int KDIM   = 4096;   // K
constexpr int NOUT   = 11008;  // N

constexpr int BM  = 128;
constexpr int BN  = 256;
constexpr int BK  = 32;
constexpr int NKT = KDIM / BK;         // 128 K-steps
constexpr unsigned A_TILE_BYTES = BM * BK * 2;   // 8 KB

// int (exact small value) -> bf16 pair packed in a dword; truncation is exact
// for integers in [0,127).
__device__ __forceinline__ unsigned pack_bf16(int a, int b) {
    union { float f; unsigned u; } ua, ub;
    ua.f = (float)a;
    ub.f = (float)b;
    return (ua.u >> 16) | (ub.u & 0xFFFF0000u);
}

// Tensor Data Mover: DMA a BM x BK bf16 tile of x[TOKENS][KDIM] into LDS.
// D# per CDNA5 ISA ch.8: group0 = {count, lds_addr, global_addr, type=2},
// group1 = {data_size=2B, tensor dims, tile dims, stride}. 2D tensor ->
// remaining groups zero. This toolchain exposes the 6-arg builtin form.
__device__ __forceinline__ void tdm_load_x_tile(const __hip_bfloat16* x,
                                                int m0, int k0,
                                                unsigned lds_byte_off) {
    const unsigned long long gaddr =
        (unsigned long long)(uintptr_t)x +
        (((unsigned long long)m0 * KDIM + (unsigned)k0) * 2ull);

    v4u g0;
    g0.x = 1u;                                   // count=1, user desc
    g0.y = lds_byte_off;                         // lds_addr
    g0.z = (unsigned)gaddr;                      // global_addr[31:0]
    g0.w = (unsigned)((gaddr >> 32) & 0x01FFFFFFu) | (2u << 30);  // type=2

    v8i g1;
    g1[0] = 1 << 16;                             // data_size=1 -> 2 bytes
    g1[1] = (KDIM & 0xFFFF) << 16;               // tensor_dim0 lo16
    g1[2] = (KDIM >> 16) | ((TOKENS & 0xFFFF) << 16);  // dim0 hi / dim1 lo
    g1[3] = (TOKENS >> 16) | (BK << 16);         // dim1 hi / tile_dim0 = 32
    g1[4] = BM;                                  // tile_dim1 = 128, tile_dim2 = 0
    g1[5] = KDIM;                                // tensor_dim0_stride lo32
    g1[6] = 0;                                   // stride hi / dim1_stride lo
    g1[7] = 0;                                   // dim1_stride hi

    v4i z4 = {0, 0, 0, 0};
    v8i z8 = {0, 0, 0, 0, 0, 0, 0, 0};
    __builtin_amdgcn_tensor_load_to_lds(g0, g1, z4, z4, z8, 0);
}

__global__ __launch_bounds__(256)
void int8_linear_wmma_bf16(const __hip_bfloat16* __restrict__ x,
                           const int* __restrict__ wq,
                           const float* __restrict__ scale,
                           const __hip_bfloat16* __restrict__ bias,
                           __hip_bfloat16* __restrict__ out)
{
    __shared__ __align__(16) __bf16 lds_a[2][BM * BK];   // 2 x 8 KB  (x tile)
    __shared__ __align__(16) __bf16 lds_b[2][BN * BK];   // 2 x 16 KB (dequant W)

    const int tid  = threadIdx.x;
    const int lane = tid & 31;
    const int wave = tid >> 5;
    const int wm   = wave >> 2;    // 0..1 : 64 M-rows per wave
    const int wn   = wave & 3;     // 0..3 : 64 N-cols per wave

    const int m0 = blockIdx.y * BM;
    const int n0 = blockIdx.x * BN;

    const float s = scale[0];

    v8f acc[4][4] = {};            // 4x4 tiles of 16x16 f32 (64x64 per wave)

    // Weight staging: thread t owns W row (n0 + t), 32 int32 per K-step.
    const int* wg = wq + (size_t)(n0 + tid) * KDIM;

    // Fragment addressing (contiguous k order: lanes 0-15 K-low, 16-31 K-high).
    const int frow = lane & 15;
    const int fkh  = (lane >> 4) * 16;

    const unsigned lds_a_off = (unsigned)(uintptr_t)(&lds_a[0][0]);

    // ---------------- prologue: stage tile 0 ----------------
    if (wave == 0)
        tdm_load_x_tile(x, m0, 0, lds_a_off);
    {
        const int4* wsrc = (const int4*)(wg + 0);
        int4 q[8];
#pragma unroll
        for (int p = 0; p < 8; ++p) q[p] = wsrc[p];
        uint4* lb = (uint4*)(&lds_b[0][tid * BK]);
#pragma unroll
        for (int p = 0; p < 4; ++p) {
            uint4 c;
            c.x = pack_bf16(q[2*p].x,   q[2*p].y);
            c.y = pack_bf16(q[2*p].z,   q[2*p].w);
            c.z = pack_bf16(q[2*p+1].x, q[2*p+1].y);
            c.w = pack_bf16(q[2*p+1].z, q[2*p+1].w);
            lb[p] = c;
        }
    }
    if (wave == 0)
        __builtin_amdgcn_s_wait_tensorcnt(0);
    __syncthreads();

    // ---------------- main K loop, double buffered ----------------
    for (int kt = 0; kt < NKT; ++kt) {
        const int cb = kt & 1;
        const int nb = cb ^ 1;
        const bool have_next = (kt + 1) < NKT;

        int4 q[8];
        if (have_next) {
            if (wave == 0)
                tdm_load_x_tile(x, m0, (kt + 1) * BK,
                                lds_a_off + (unsigned)nb * A_TILE_BYTES);
            const int4* wsrc = (const int4*)(wg + (kt + 1) * BK);
#pragma unroll
            for (int p = 0; p < 8; ++p) q[p] = wsrc[p];
        }

        // ---- fragment loads from current buffer ----
        v16bf afrag[4], bfrag[4];
#pragma unroll
        for (int i = 0; i < 4; ++i)
            afrag[i] = *(const v16bf*)(&lds_a[cb][(wm * 64 + i * 16 + frow) * BK + fkh]);
#pragma unroll
        for (int j = 0; j < 4; ++j)
            bfrag[j] = *(const v16bf*)(&lds_b[cb][(wn * 64 + j * 16 + frow) * BK + fkh]);

        // ---- 16x v_wmma_f32_16x16x32_bf16 ----
#pragma unroll
        for (int i = 0; i < 4; ++i)
#pragma unroll
            for (int j = 0; j < 4; ++j)
                acc[i][j] = __builtin_amdgcn_wmma_f32_16x16x32_bf16(
                    false, afrag[i], false, bfrag[j],
                    (short)0, acc[i][j], false, false);

        // ---- dequantize + store next weight tile ----
        if (have_next) {
            uint4* lb = (uint4*)(&lds_b[nb][tid * BK]);
#pragma unroll
            for (int p = 0; p < 4; ++p) {
                uint4 c;
                c.x = pack_bf16(q[2*p].x,   q[2*p].y);
                c.y = pack_bf16(q[2*p].z,   q[2*p].w);
                c.z = pack_bf16(q[2*p+1].x, q[2*p+1].y);
                c.w = pack_bf16(q[2*p+1].z, q[2*p+1].w);
                lb[p] = c;
            }
            if (wave == 0)
                __builtin_amdgcn_s_wait_tensorcnt(0);
        }
        __syncthreads();
    }

    // ---------------- epilogue: out = acc * scale + bias ----------------
    // C/D layout: VGPR r, lane<16 -> M=r, N=lane; lane>=16 -> M=8+r, N=lane-16.
    const int msel = (lane >> 4) * 8;
    const int ncol = lane & 15;
#pragma unroll
    for (int j = 0; j < 4; ++j) {
        const int n = n0 + wn * 64 + j * 16 + ncol;
        const float bv = __bfloat162float(bias[n]);
#pragma unroll
        for (int i = 0; i < 4; ++i) {
#pragma unroll
            for (int r = 0; r < 8; ++r) {
                const int m = m0 + wm * 64 + i * 16 + msel + r;
                out[(size_t)m * NOUT + n] = __float2bfloat16(acc[i][j][r] * s + bv);
            }
        }
    }
}

extern "C" void kernel_launch(void* const* d_in, const int* in_sizes, int n_in,
                              void* d_out, int out_size, void* d_ws, size_t ws_size,
                              hipStream_t stream) {
    (void)in_sizes; (void)n_in; (void)d_ws; (void)ws_size; (void)out_size;
    const __hip_bfloat16* x     = (const __hip_bfloat16*)d_in[0];
    const int*            wq    = (const int*)d_in[1];
    const float*          scale = (const float*)d_in[2];
    const __hip_bfloat16* bias  = (const __hip_bfloat16*)d_in[3];
    __hip_bfloat16*       out   = (__hip_bfloat16*)d_out;

    dim3 grid(NOUT / BN, TOKENS / BM);   // 43 x 32 blocks
    int8_linear_wmma_bf16<<<grid, 256, 0, stream>>>(x, wq, scale, bias, out);
}
